// GSR_7971459301537
// MI455X (gfx1250) — matrix operand order repaired and checked by reading, not compiled, not run on hardware
//
#include <hip/hip_runtime.h>

// ---------------------------------------------------------------------------
// Problem constants (match reference)
// ---------------------------------------------------------------------------
#define N      16384
#define D      256
#define E      524288
#define K_TOP  4096
#define BIG    99.0f
#define EPS    1e-8f

#define NBINS  4096
#define VMIN   (-1.5f)
#define VMAX   (1.5f)
#define BINW   ((VMAX - VMIN) / (float)NBINS)
#define BINSCALE ((float)NBINS / (VMAX - VMIN))
#define CAP    262144              // candidate buffer capacity

typedef __bf16 bf16_t;
typedef __attribute__((ext_vector_type(8)))  bf16_t v8bf;
typedef __attribute__((ext_vector_type(16))) bf16_t v16bf;
typedef __attribute__((ext_vector_type(8)))  float  v8f;

// ---------------------------------------------------------------------------
// Workspace layout (bytes). All offsets 256B aligned.
// ---------------------------------------------------------------------------
#define O_BF      ((size_t)0)                         // N*D ushort  = 8 MiB
#define O_BS      (O_BF + (size_t)N * D * 2)          // N*D ushort  = 8 MiB
#define O_NF      (O_BS + (size_t)N * D * 2)          // N float
#define O_NS      (O_NF + (size_t)N * 4)              // N float
#define O_BITMAP  (O_NS + (size_t)N * 4)              // N*N/8 bytes = 32 MiB
#define O_HISTA   (O_BITMAP + (size_t)N * (N / 8))    // NBINS int
#define O_HISTR   (O_HISTA + (size_t)NBINS * 4)       // NBINS int
#define O_CNT     (O_HISTR + (size_t)NBINS * 4)       // 2 int (padded 256)
#define O_THR     (O_CNT + 256)                        // 2 float (padded 256)
#define O_CANDAV  (O_THR + 256)                        // CAP float
#define O_CANDAI  (O_CANDAV + (size_t)CAP * 4)         // CAP uint
#define O_CANDRV  (O_CANDAI + (size_t)CAP * 4)         // CAP float
#define O_CANDRI  (O_CANDRV + (size_t)CAP * 4)         // CAP uint
#define O_EDGESIM (O_CANDRI + (size_t)CAP * 4)         // E float
#define ZERO_BYTES (O_THR - O_BITMAP)                  // bitmap+hists+counters

// ---------------------------------------------------------------------------
// Helpers
// ---------------------------------------------------------------------------
__device__ inline unsigned short f2bf(float f) {   // round-to-nearest-even
    unsigned u = __float_as_uint(f);
    u += 0x7FFFu + ((u >> 16) & 1u);
    return (unsigned short)(u >> 16);
}

__device__ inline int vbin(float v) {
    int b = (int)((v - VMIN) * BINSCALE);
    return b < 0 ? 0 : (b > NBINS - 1 ? NBINS - 1 : b);
}

__device__ inline v8f wmma_bf16(v16bf a, v16bf b, v8f c) {
    return __builtin_amdgcn_wmma_f32_16x16x32_bf16(
        false, a, false, b, (short)0, c, false, false);
}

// A fragment (16x32, 16-bit): lane holds row (lane&15); elems 0..15 -> K =
// 16*(lane>>4)+e  (contiguous 32B).
__device__ inline v16bf load_a(const unsigned short* base, int row0, int kk, int lane) {
    int r  = row0 + (lane & 15);
    int kb = kk + ((lane >> 4) << 4);
    return *reinterpret_cast<const v16bf*>(base + (size_t)r * D + kb);
}

// B fragment (32x16, 16-bit): lane holds col (lane&15); elems 0..15 -> K =
// 16*(lane>>4)+e  (contiguous 32B).  B = (emb tile)^T so we read rows of emb.
__device__ inline v16bf load_b(const unsigned short* base, int row0, int kk, int lane) {
    int r  = row0 + (lane & 15);
    int kb = kk + ((lane >> 4) << 4);
    return *reinterpret_cast<const v16bf*>(base + (size_t)r * D + kb);
}

// ---------------------------------------------------------------------------
// Kernel 0: zero scratch region (bitmap + histograms + counters)
// ---------------------------------------------------------------------------
__global__ void k_zero(unsigned* p, size_t nwords) {
    size_t stride = (size_t)gridDim.x * blockDim.x;
    for (size_t i = (size_t)blockIdx.x * blockDim.x + threadIdx.x; i < nwords; i += stride)
        p[i] = 0u;
}

// ---------------------------------------------------------------------------
// Kernel 1: norms + f32 -> bf16 pack. One wave per row.
// ---------------------------------------------------------------------------
__global__ __launch_bounds__(256) void k_prep(const float* __restrict__ F,
                                              const float* __restrict__ S,
                                              unsigned short* __restrict__ bF,
                                              unsigned short* __restrict__ bS,
                                              float* __restrict__ nF,
                                              float* __restrict__ nS) {
    int lane = threadIdx.x & 31;
    int row  = blockIdx.x * 8 + (threadIdx.x >> 5);
    if (row >= N) return;
    const float* fr = F + (size_t)row * D + lane * 8;
    const float* sr = S + (size_t)row * D + lane * 8;
    unsigned short* fo = bF + (size_t)row * D + lane * 8;
    unsigned short* so = bS + (size_t)row * D + lane * 8;
    float fs = 0.f, ss = 0.f;
#pragma unroll
    for (int i = 0; i < 8; ++i) {
        float x = fr[i]; fs += x * x; fo[i] = f2bf(x);
        float y = sr[i]; ss += y * y; so[i] = f2bf(y);
    }
#pragma unroll
    for (int off = 16; off; off >>= 1) {
        fs += __shfl_xor(fs, off, 32);
        ss += __shfl_xor(ss, off, 32);
    }
    if (lane == 0) { nF[row] = sqrtf(fs); nS[row] = sqrtf(ss); }
}

// ---------------------------------------------------------------------------
// Kernel 2: edge bitmap (idempotent atomicOr)
// ---------------------------------------------------------------------------
__global__ void k_bitmap(const int* __restrict__ er, const int* __restrict__ ec,
                         unsigned* __restrict__ bitmap) {
    int e = blockIdx.x * blockDim.x + threadIdx.x;
    if (e >= E) return;
    unsigned pos = (unsigned)er[e] * (unsigned)N + (unsigned)ec[e];
    atomicOr(&bitmap[pos >> 5], 1u << (pos & 31));
}

// ---------------------------------------------------------------------------
// Kernel 3: per-edge sims (one wave per edge) + low-side histogram
// ---------------------------------------------------------------------------
__global__ __launch_bounds__(256) void k_edge_sim(const float* __restrict__ F,
                                                  const float* __restrict__ S,
                                                  const float* __restrict__ nF,
                                                  const float* __restrict__ nS,
                                                  const int* __restrict__ er,
                                                  const int* __restrict__ ec,
                                                  float* __restrict__ edgeSim,
                                                  int* __restrict__ histR) {
    int lane = threadIdx.x & 31;
    int e    = blockIdx.x * 8 + (threadIdx.x >> 5);
    if (e >= E) return;
    int r = er[e], c = ec[e];
    const float* fr = F + (size_t)r * D; const float* fc = F + (size_t)c * D;
    const float* sr = S + (size_t)r * D; const float* sc = S + (size_t)c * D;
    float fd = 0.f, sd = 0.f;
#pragma unroll
    for (int i = 0; i < 8; ++i) {
        int d = lane + 32 * i;
        fd += fr[d] * fc[d];
        sd += sr[d] * sc[d];
    }
#pragma unroll
    for (int off = 16; off; off >>= 1) {
        fd += __shfl_xor(fd, off, 32);
        sd += __shfl_xor(sd, off, 32);
    }
    if (lane == 0) {
        float v = 0.5f * fd / fmaxf(nF[r] * nF[c], EPS)
                + 0.5f * sd / fmaxf(nS[r] * nS[c], EPS);
        edgeSim[e] = v;
        atomicAdd(&histR[vbin(v)], 1);
    }
}

// ---------------------------------------------------------------------------
// Kernel 4: WMMA GEMM over sim matrix.
//   block = 256 thr (8 waves), tile 64(M) x 128(N); wave tile 32x32 (2x2 frags)
//   mode 0: histogram of add-eligible values (non-edge, non-diag)
//   mode 1: collect candidates with val >= *thr
// ---------------------------------------------------------------------------
__global__ __launch_bounds__(256) void k_gemm(const unsigned short* __restrict__ bF,
                                              const unsigned short* __restrict__ bS,
                                              const float* __restrict__ nF,
                                              const float* __restrict__ nS,
                                              const unsigned* __restrict__ bitmap,
                                              int mode,
                                              const float* __restrict__ tptr,
                                              int* __restrict__ histA,
                                              float* __restrict__ candV,
                                              unsigned* __restrict__ candI,
                                              int* __restrict__ counter) {
    __shared__ int lhist[NBINS];
    if (mode == 0) {
        for (int i = threadIdx.x; i < NBINS; i += 256) lhist[i] = 0;
        __syncthreads();
    }
    const int lane  = threadIdx.x & 31;
    const int wave  = threadIdx.x >> 5;
    const int tileM = blockIdx.y * 64  + (wave >> 2) * 32;
    const int tileN = blockIdx.x * 128 + (wave & 3) * 32;
    const float thr = (mode == 1) ? tptr[0] : 0.f;

    v8f cF[2][2], cS[2][2];
#pragma unroll
    for (int i = 0; i < 2; ++i)
#pragma unroll
        for (int j = 0; j < 2; ++j) { v8f z = {}; cF[i][j] = z; cS[i][j] = z; }

#pragma unroll
    for (int kk = 0; kk < D; kk += 32) {
        v16bf a0 = load_a(bF, tileM,      kk, lane);
        v16bf a1 = load_a(bF, tileM + 16, kk, lane);
        v16bf b0 = load_b(bF, tileN,      kk, lane);
        v16bf b1 = load_b(bF, tileN + 16, kk, lane);
        cF[0][0] = wmma_bf16(a0, b0, cF[0][0]);
        cF[0][1] = wmma_bf16(a0, b1, cF[0][1]);
        cF[1][0] = wmma_bf16(a1, b0, cF[1][0]);
        cF[1][1] = wmma_bf16(a1, b1, cF[1][1]);
    }
#pragma unroll
    for (int kk = 0; kk < D; kk += 32) {
        v16bf a0 = load_a(bS, tileM,      kk, lane);
        v16bf a1 = load_a(bS, tileM + 16, kk, lane);
        v16bf b0 = load_b(bS, tileN,      kk, lane);
        v16bf b1 = load_b(bS, tileN + 16, kk, lane);
        cS[0][0] = wmma_bf16(a0, b0, cS[0][0]);
        cS[0][1] = wmma_bf16(a0, b1, cS[0][1]);
        cS[1][0] = wmma_bf16(a1, b0, cS[1][0]);
        cS[1][1] = wmma_bf16(a1, b1, cS[1][1]);
    }

    // C/D layout: elem j -> m_local = j + 8*(lane>>4), n_local = lane&15
#pragma unroll
    for (int fi = 0; fi < 2; ++fi)
#pragma unroll
        for (int fj = 0; fj < 2; ++fj)
#pragma unroll
            for (int j = 0; j < 8; ++j) {
                int m = tileM + fi * 16 + j + ((lane >> 4) << 3);
                int n = tileN + fj * 16 + (lane & 15);
                float fd = cF[fi][fj][j];
                float sd = cS[fi][fj][j];
                float val = 0.5f * fd / fmaxf(nF[m] * nF[n], EPS)
                          + 0.5f * sd / fmaxf(nS[m] * nS[n], EPS);
                unsigned pos = (unsigned)m * (unsigned)N + (unsigned)n;
                bool edge = (bitmap[pos >> 5] >> (pos & 31)) & 1u;
                if (!edge && m != n) {
                    if (mode == 0) {
                        atomicAdd(&lhist[vbin(val)], 1);
                    } else if (val >= thr) {
                        int id = atomicAdd(counter, 1);
                        if (id < CAP) {
                            candV[id] = val;
                            candI[id] = ((unsigned)m << 14) | (unsigned)n;
                        }
                    }
                }
            }

    if (mode == 0) {
        __syncthreads();
        for (int i = threadIdx.x; i < NBINS; i += 256)
            if (lhist[i]) atomicAdd(&histA[i], lhist[i]);
    }
}

// ---------------------------------------------------------------------------
// Kernel 5: derive threshold from histogram (single thread; 4096 iters)
// ---------------------------------------------------------------------------
__global__ void k_thresh(const int* __restrict__ hist, float* __restrict__ out,
                         int fromTop) {
    long long cum = 0;
    if (fromTop) {
        for (int b = NBINS - 1; b >= 0; --b) {
            cum += hist[b];
            if (cum >= K_TOP) { out[0] = VMIN + b * BINW; return; }
        }
        out[0] = VMIN;
    } else {
        for (int b = 0; b < NBINS; ++b) {
            cum += hist[b];
            if (cum >= K_TOP) { out[0] = VMIN + (b + 1) * BINW; return; }
        }
        out[0] = VMAX;
    }
}

// ---------------------------------------------------------------------------
// Kernel 6: collect rm candidates (values stored NEGATED for max-select)
// ---------------------------------------------------------------------------
__global__ void k_collect_rm(const float* __restrict__ edgeSim,
                             const int* __restrict__ er, const int* __restrict__ ec,
                             const float* __restrict__ tptr,
                             float* __restrict__ candV, unsigned* __restrict__ candI,
                             int* __restrict__ counter) {
    int e = blockIdx.x * blockDim.x + threadIdx.x;
    if (e >= E) return;
    float v = edgeSim[e];
    if (v <= tptr[0]) {
        int id = atomicAdd(counter, 1);
        if (id < CAP) {
            candV[id] = -v;
            candI[id] = ((unsigned)er[e] << 14) | (unsigned)ec[e];
        }
    }
}

// ---------------------------------------------------------------------------
// Kernel 7: exact top-K by iterative arg-max (deterministic tie-break on idx)
// ---------------------------------------------------------------------------
__global__ __launch_bounds__(1024) void k_select(float* __restrict__ vals,
                                                 const unsigned* __restrict__ idxs,
                                                 const int* __restrict__ counter,
                                                 float* __restrict__ outV,
                                                 int* __restrict__ outI,
                                                 int negate, float pad) {
    __shared__ float    sv[1024];
    __shared__ unsigned si[1024];
    __shared__ int      sj[1024];
    int cnt = *counter; if (cnt > CAP) cnt = CAP;
    const int t = threadIdx.x;
    for (int it = 0; it < K_TOP; ++it) {
        float bv = -3.0e38f; unsigned bi = 0xFFFFFFFFu; int bj = -1;
        for (int i = t; i < cnt; i += 1024) {
            float v = vals[i]; unsigned p = idxs[i];
            if (v > bv || (v == bv && p < bi)) { bv = v; bi = p; bj = i; }
        }
        sv[t] = bv; si[t] = bi; sj[t] = bj;
        __syncthreads();
        for (int s = 512; s; s >>= 1) {
            if (t < s) {
                float v2 = sv[t + s]; unsigned p2 = si[t + s];
                if (v2 > sv[t] || (v2 == sv[t] && p2 < si[t])) {
                    sv[t] = v2; si[t] = p2; sj[t] = sj[t + s];
                }
            }
            __syncthreads();
        }
        if (t == 0) {
            if (sj[0] >= 0 && sv[0] > -1.0e38f) {
                outV[it]         = negate ? -sv[0] : sv[0];
                outI[2 * it]     = (int)(si[0] >> 14);
                outI[2 * it + 1] = (int)(si[0] & 16383u);
                vals[sj[0]] = -3.0e38f;          // consume
            } else {
                outV[it] = pad; outI[2 * it] = 0; outI[2 * it + 1] = 0;
            }
        }
        __threadfence_block();
        __syncthreads();
    }
}

// ---------------------------------------------------------------------------
// Launch
// ---------------------------------------------------------------------------
extern "C" void kernel_launch(void* const* d_in, const int* in_sizes, int n_in,
                              void* d_out, int out_size, void* d_ws, size_t ws_size,
                              hipStream_t stream) {
    const float* emb_F = (const float*)d_in[0];
    const float* emb_S = (const float*)d_in[1];
    const int*   er    = (const int*)d_in[2];
    const int*   ec    = (const int*)d_in[3];

    char* ws = (char*)d_ws;
    unsigned short* bF      = (unsigned short*)(ws + O_BF);
    unsigned short* bS      = (unsigned short*)(ws + O_BS);
    float*          nF      = (float*)(ws + O_NF);
    float*          nS      = (float*)(ws + O_NS);
    unsigned*       bitmap  = (unsigned*)(ws + O_BITMAP);
    int*            histA   = (int*)(ws + O_HISTA);
    int*            histR   = (int*)(ws + O_HISTR);
    int*            cnt     = (int*)(ws + O_CNT);
    float*          thr     = (float*)(ws + O_THR);
    float*          candAv  = (float*)(ws + O_CANDAV);
    unsigned*       candAi  = (unsigned*)(ws + O_CANDAI);
    float*          candRv  = (float*)(ws + O_CANDRV);
    unsigned*       candRi  = (unsigned*)(ws + O_CANDRI);
    float*          edgeSim = (float*)(ws + O_EDGESIM);

    // 0) zero bitmap + histograms + counters
    k_zero<<<4096, 256, 0, stream>>>(bitmap, ZERO_BYTES / 4);
    // 1) norms + bf16 pack
    k_prep<<<N / 8, 256, 0, stream>>>(emb_F, emb_S, bF, bS, nF, nS);
    // 2) edge bitmap
    k_bitmap<<<E / 256, 256, 0, stream>>>(er, ec, bitmap);
    // 3) per-edge sims + rm histogram
    k_edge_sim<<<E / 8, 256, 0, stream>>>(emb_F, emb_S, nF, nS, er, ec,
                                          edgeSim, histR);
    // 4) GEMM pass 1: add-side histogram
    k_gemm<<<dim3(N / 128, N / 64), 256, 0, stream>>>(
        bF, bS, nF, nS, bitmap, 0, thr, histA, candAv, candAi, &cnt[0]);
    // 5) thresholds
    k_thresh<<<1, 1, 0, stream>>>(histA, &thr[0], 1);
    k_thresh<<<1, 1, 0, stream>>>(histR, &thr[1], 0);
    // 6) GEMM pass 2: collect add candidates
    k_gemm<<<dim3(N / 128, N / 64), 256, 0, stream>>>(
        bF, bS, nF, nS, bitmap, 1, &thr[0], histA, candAv, candAi, &cnt[0]);
    // 7) collect rm candidates
    k_collect_rm<<<E / 256, 256, 0, stream>>>(edgeSim, er, ec, &thr[1],
                                              candRv, candRi, &cnt[1]);
    // 8) exact top-K selects -> output layout: rm_v | rm_i | add_v | add_i
    float* out = (float*)d_out;
    k_select<<<1, 1024, 0, stream>>>(candRv, candRi, &cnt[1],
                                     out, (int*)(out + K_TOP), 1, BIG);
    k_select<<<1, 1024, 0, stream>>>(candAv, candAi, &cnt[0],
                                     out + 3 * K_TOP, (int*)(out + 4 * K_TOP),
                                     0, -BIG);
}
